// Qwen3MoeAfdMlpLayer_58317065945110
// MI455X (gfx1250) — matrix operand order repaired and checked by reading, not compiled, not run on hardware
//
#include <hip/hip_runtime.h>

// Problem constants (from reference)
#define TT   1024
#define HHID 1024
#define EE   64
#define KTOP 8
#define FF   768
#define SS   (TT * KTOP)      // 8192 routed rows
#define CAP  512              // per-expert capacity (4x average), matches reference
#define LDA  40               // LDS row stride in bf16 elems (80B: 16B-aligned, bank-friendly)

typedef __attribute__((ext_vector_type(16))) __bf16 v16bf;
typedef __attribute__((ext_vector_type(8)))  float  v8f;
typedef __attribute__((ext_vector_type(4)))  float  f32x4;
typedef __attribute__((ext_vector_type(4)))  __bf16 bf16x4;

union Frag16 { v16bf v; uint4 q[2]; };

// fp32x4 -> bf16x4 (RNE): lowers to v_cvt_pk_bf16_f32 pairs
__device__ __forceinline__ void st4bf(unsigned short* d, f32x4 v) {
  union { bf16x4 b; uint2 u; } cv;
  cv.b = __builtin_convertvector(v, bf16x4);
  *(uint2*)d = cv.u;
}
__device__ __forceinline__ unsigned short f2bf1(float f) {
  union { __bf16 b; unsigned short s; } cv;
  cv.b = (__bf16)f;
  return cv.s;
}

// A fragment (16x32 bf16): lane m = lane&15; half=lane>>4 picks K {0..7,16..23} / {8..15,24..31}
__device__ __forceinline__ v16bf ldsA(const unsigned short* s, int row, int half) {
  Frag16 r;
  r.q[0] = *(const uint4*)(s + row * LDA + half * 8);
  r.q[1] = *(const uint4*)(s + row * LDA + 16 + half * 8);
  return r.v;
}
// B fragment (32x16 bf16): lane n = lane&15; half=lane>>4 picks K {0..7,16..23} / {8..15,24..31}
__device__ __forceinline__ v16bf ldsB(const unsigned short* s, int col, int half) {
  Frag16 r;
  r.q[0] = *(const uint4*)(s + col * LDA + half * 8);
  r.q[1] = *(const uint4*)(s + col * LDA + 16 + half * 8);
  return r.v;
}

// ---------------- routing: scatter (token, weight) into per-expert slots ----------------
__global__ void route_k(const int* __restrict__ ids, const float* __restrict__ tw,
                        int* __restrict__ counts, int* __restrict__ stok,
                        float* __restrict__ sw) {
  int s = blockIdx.x * blockDim.x + threadIdx.x;
  if (s >= SS) return;
  int e = ids[s];
  int p = atomicAdd(&counts[e], 1);
  if (p < CAP) {
    stok[e * CAP + p] = s / KTOP;
    sw[e * CAP + p]   = tw[s];
  }
}

// ---------------- GEMM1 + SwiGLU: act[e,c,f] = silu(x@w1g^T) * (x@w1u^T) ----------------
__global__ __launch_bounds__(256) void gemm1_k(const float* __restrict__ x,
                                               const float* __restrict__ w1,
                                               const int* __restrict__ counts,
                                               const int* __restrict__ stok,
                                               unsigned short* __restrict__ act) {
  __shared__ unsigned short As[2][128 * LDA];   // ping-pong buffers
  __shared__ unsigned short Gs[2][64 * LDA];
  __shared__ unsigned short Us[2][64 * LDA];

  const int e   = blockIdx.z;
  const int n_e = min(counts[e], CAP);
  const int m0  = blockIdx.y * 128;
  if (m0 >= n_e) return;                 // uniform exit: EXEC stays all-1 for WMMA
  const int f0  = blockIdx.x * 64;

  const int tid  = threadIdx.x;
  const int lane = tid & 31;
  const int wv   = tid >> 5;             // 8 waves: 4(M) x 2(N)
  const int wm   = wv >> 1;
  const int wn   = wv & 1;
  const int hl   = lane >> 4;
  const int ln   = lane & 15;

  // A loader: 2 threads/row, 16 fp32 each
  const int arow = tid >> 1;
  const int acol = (tid & 1) * 16;
  const float* asrc = nullptr;
  { int c = m0 + arow;
    if (c < n_e) asrc = x + (size_t)stok[e * CAP + c] * HHID; }

  // B loaders: 4 threads/row, 8 fp32 each (gate + up)
  const int brow = tid >> 2;
  const int bcol = (tid & 3) * 8;
  const float* gsrc = w1 + ((size_t)e * (2 * FF) + (f0 + brow)) * HHID;
  const float* usrc = w1 + ((size_t)e * (2 * FF) + (FF + f0 + brow)) * HHID;

  // Zero-fill invalid token rows ONCE in both buffers (never overwritten later)
  if (!asrc) {
    uint2 z = make_uint2(0u, 0u);
#pragma unroll
    for (int p = 0; p < 2; ++p) {
      unsigned short* d = As[p] + arow * LDA + acol;
      *(uint2*)(d + 0) = z; *(uint2*)(d + 4) = z;
      *(uint2*)(d + 8) = z; *(uint2*)(d + 12) = z;
    }
  }

  // register staging of global loads (one k-step ahead of LDS commit)
  f32x4 ra0, ra1, ra2, ra3, rg0, rg1, ru0, ru1;
  auto stage = [&](int k0) {
    if (asrc) {
      ra0 = *(const f32x4*)(asrc + k0 + acol + 0);
      ra1 = *(const f32x4*)(asrc + k0 + acol + 4);
      ra2 = *(const f32x4*)(asrc + k0 + acol + 8);
      ra3 = *(const f32x4*)(asrc + k0 + acol + 12);
    }
    rg0 = *(const f32x4*)(gsrc + k0 + bcol + 0);
    rg1 = *(const f32x4*)(gsrc + k0 + bcol + 4);
    ru0 = *(const f32x4*)(usrc + k0 + bcol + 0);
    ru1 = *(const f32x4*)(usrc + k0 + bcol + 4);
    if (k0 + 96 <= HHID) {               // L2 prefetch further ahead
      __builtin_prefetch(gsrc + k0 + 64, 0, 3);
      __builtin_prefetch(usrc + k0 + 64, 0, 3);
    }
  };
  // convert + commit staged registers into buffer p
  auto commit = [&](int p) {
    if (asrc) {
      unsigned short* d = As[p] + arow * LDA + acol;
      st4bf(d + 0, ra0); st4bf(d + 4, ra1);
      st4bf(d + 8, ra2); st4bf(d + 12, ra3);
    }
    unsigned short* dg = Gs[p] + brow * LDA + bcol;
    st4bf(dg + 0, rg0); st4bf(dg + 4, rg1);
    unsigned short* du = Us[p] + brow * LDA + bcol;
    st4bf(du + 0, ru0); st4bf(du + 4, ru1);
  };

  stage(0);
  commit(0);
  stage(32);
  __syncthreads();

  v8f accG[2][2] = {}; v8f accU[2][2] = {};

  for (int k0 = 0; k0 < HHID; k0 += 32) {
    const int p = (k0 >> 5) & 1;

    v16bf a0  = ldsA(As[p], wm * 32 + ln,      hl);
    v16bf a1  = ldsA(As[p], wm * 32 + 16 + ln, hl);
    v16bf bg0 = ldsB(Gs[p], wn * 32 + ln,      hl);
    v16bf bg1 = ldsB(Gs[p], wn * 32 + 16 + ln, hl);
    v16bf bu0 = ldsB(Us[p], wn * 32 + ln,      hl);
    v16bf bu1 = ldsB(Us[p], wn * 32 + 16 + ln, hl);

    // commit next tile into the other buffer and stage the one after
    if (k0 + 32 < HHID) {
      commit(p ^ 1);
      if (k0 + 64 < HHID) stage(k0 + 64);
    }

    accG[0][0] = __builtin_amdgcn_wmma_f32_16x16x32_bf16(false, a0, false, bg0, (short)0, accG[0][0], false, false);
    accG[0][1] = __builtin_amdgcn_wmma_f32_16x16x32_bf16(false, a0, false, bg1, (short)0, accG[0][1], false, false);
    accG[1][0] = __builtin_amdgcn_wmma_f32_16x16x32_bf16(false, a1, false, bg0, (short)0, accG[1][0], false, false);
    accG[1][1] = __builtin_amdgcn_wmma_f32_16x16x32_bf16(false, a1, false, bg1, (short)0, accG[1][1], false, false);
    accU[0][0] = __builtin_amdgcn_wmma_f32_16x16x32_bf16(false, a0, false, bu0, (short)0, accU[0][0], false, false);
    accU[0][1] = __builtin_amdgcn_wmma_f32_16x16x32_bf16(false, a0, false, bu1, (short)0, accU[0][1], false, false);
    accU[1][0] = __builtin_amdgcn_wmma_f32_16x16x32_bf16(false, a1, false, bu0, (short)0, accU[1][0], false, false);
    accU[1][1] = __builtin_amdgcn_wmma_f32_16x16x32_bf16(false, a1, false, bu1, (short)0, accU[1][1], false, false);

    __syncthreads();   // single barrier per step: publishes buf[p^1], retires buf[p]
  }

  // fused SiLU(g)*u epilogue -> bf16 act
  unsigned short* abase = act + (size_t)e * CAP * FF;
#pragma unroll
  for (int mi = 0; mi < 2; ++mi)
#pragma unroll
    for (int ni = 0; ni < 2; ++ni) {
      const int cb = m0 + wm * 32 + mi * 16 + hl * 8;
      const int fb = f0 + wn * 32 + ni * 16 + ln;
      v8f g = accG[mi][ni], u = accU[mi][ni];
#pragma unroll
      for (int r = 0; r < 8; ++r) {
        float gv = g[r];
        float av = (gv / (1.0f + __expf(-gv))) * u[r];
        abase[(size_t)(cb + r) * FF + fb] = f2bf1(av);
      }
    }
}

// ---------------- GEMM2 + weighted combine: out[t,:] += tw * (act @ w2^T) ----------------
__global__ __launch_bounds__(256) void gemm2_k(const unsigned short* __restrict__ act,
                                               const float* __restrict__ w2,
                                               const int* __restrict__ counts,
                                               const int* __restrict__ stok,
                                               const float* __restrict__ sw,
                                               float* __restrict__ out) {
  __shared__ unsigned short As[2][128 * LDA];
  __shared__ unsigned short Ws[2][64 * LDA];
  __shared__ int   Ttok[128];
  __shared__ float Wgt[128];

  const int e   = blockIdx.z;
  const int n_e = min(counts[e], CAP);
  const int m0  = blockIdx.y * 128;
  if (m0 >= n_e) return;
  const int h0  = blockIdx.x * 64;

  const int tid  = threadIdx.x;
  const int lane = tid & 31;
  const int wv   = tid >> 5;
  const int wm   = wv >> 1;
  const int wn   = wv & 1;
  const int hl   = lane >> 4;
  const int ln   = lane & 15;

  if (tid < 128) {
    int c = m0 + tid;
    if (c < n_e) { Ttok[tid] = stok[e * CAP + c]; Wgt[tid] = sw[e * CAP + c]; }
    else         { Ttok[tid] = 0;                 Wgt[tid] = 0.0f; }
  }

  const int arow = tid >> 1;
  const int acol = (tid & 1) * 16;
  const uint4* asrc = (const uint4*)(act + ((size_t)e * CAP + m0 + arow) * FF);

  const int brow = tid >> 2;
  const int bcol = (tid & 3) * 8;
  const float* wsrc = w2 + ((size_t)e * HHID + (h0 + brow)) * FF;

  // register staging
  uint4 sa0, sa1; f32x4 rb0, rb1;
  auto stage = [&](int k0) {
    int qi = (k0 + acol) >> 3;
    sa0 = asrc[qi];
    sa1 = asrc[qi + 1];
    rb0 = *(const f32x4*)(wsrc + k0 + bcol + 0);
    rb1 = *(const f32x4*)(wsrc + k0 + bcol + 4);
    if (k0 + 96 <= FF) __builtin_prefetch(wsrc + k0 + 64, 0, 3);
  };
  auto commit = [&](int p) {
    uint4* d = (uint4*)(As[p] + arow * LDA + acol);   // act already bf16
    d[0] = sa0;
    d[1] = sa1;
    unsigned short* db = Ws[p] + brow * LDA + bcol;   // w2 fp32 -> bf16
    st4bf(db + 0, rb0); st4bf(db + 4, rb1);
  };

  stage(0);
  commit(0);
  stage(32);
  __syncthreads();

  v8f acc[2][2] = {};

  for (int k0 = 0; k0 < FF; k0 += 32) {
    const int p = (k0 >> 5) & 1;

    v16bf a0 = ldsA(As[p], wm * 32 + ln,      hl);
    v16bf a1 = ldsA(As[p], wm * 32 + 16 + ln, hl);
    v16bf b0 = ldsB(Ws[p], wn * 32 + ln,      hl);
    v16bf b1 = ldsB(Ws[p], wn * 32 + 16 + ln, hl);

    if (k0 + 32 < FF) {
      commit(p ^ 1);
      if (k0 + 64 < FF) stage(k0 + 64);
    }

    acc[0][0] = __builtin_amdgcn_wmma_f32_16x16x32_bf16(false, a0, false, b0, (short)0, acc[0][0], false, false);
    acc[0][1] = __builtin_amdgcn_wmma_f32_16x16x32_bf16(false, a0, false, b1, (short)0, acc[0][1], false, false);
    acc[1][0] = __builtin_amdgcn_wmma_f32_16x16x32_bf16(false, a1, false, b0, (short)0, acc[1][0], false, false);
    acc[1][1] = __builtin_amdgcn_wmma_f32_16x16x32_bf16(false, a1, false, b1, (short)0, acc[1][1], false, false);

    __syncthreads();
  }

  // weighted scatter-combine into out (fp32 atomics)
#pragma unroll
  for (int mi = 0; mi < 2; ++mi)
#pragma unroll
    for (int ni = 0; ni < 2; ++ni) {
      const int lrb = wm * 32 + mi * 16 + hl * 8;     // local row in tile
      const int hb  = h0 + wn * 32 + ni * 16 + ln;
      v8f y = acc[mi][ni];
#pragma unroll
      for (int r = 0; r < 8; ++r) {
        int lr = lrb + r;
        int c  = m0 + lr;
        if (c < n_e) {
          atomicAdd(out + (size_t)Ttok[lr] * HHID + hb, y[r] * Wgt[lr]);
        }
      }
    }
}

// ---------------- host launch ----------------
extern "C" void kernel_launch(void* const* d_in, const int* in_sizes, int n_in,
                              void* d_out, int out_size, void* d_ws, size_t ws_size,
                              hipStream_t stream) {
  const float* x   = (const float*)d_in[0];   // [T, H]
  const float* tw  = (const float*)d_in[1];   // [T, K]
  const int*   ids = (const int*)d_in[2];     // [T, K]
  const float* w1  = (const float*)d_in[3];   // [E, 2F, H]
  const float* w2  = (const float*)d_in[4];   // [E, H, F]
  float* out = (float*)d_out;                 // [T, H]

  // workspace layout
  int*   counts = (int*)d_ws;                       // E
  int*   stok   = counts + EE;                      // E*CAP
  float* sw     = (float*)(stok + EE * CAP);        // E*CAP
  unsigned short* act = (unsigned short*)(sw + EE * CAP); // E*CAP*F bf16 (~48 MB)

  hipMemsetAsync(counts, 0, EE * sizeof(int), stream);
  hipMemsetAsync(out, 0, (size_t)TT * HHID * sizeof(float), stream);

  route_k<<<SS / 256, 256, 0, stream>>>(ids, tw, counts, stok, sw);
  gemm1_k<<<dim3(FF / 64, CAP / 128, EE), 256, 0, stream>>>(x, w1, counts, stok, act);
  gemm2_k<<<dim3(HHID / 64, CAP / 128, EE), 256, 0, stream>>>(act, w2, counts, stok, sw, out);
}
